// BiLSTMEncoder2_46660524704398
// MI455X (gfx1250) — matrix operand order repaired and checked by reading, not compiled, not run on hardware
//
#include <hip/hip_runtime.h>

typedef __attribute__((ext_vector_type(16))) __bf16 v16bf;
typedef __attribute__((ext_vector_type(8)))  __bf16 v8bf;
typedef __attribute__((ext_vector_type(8)))  float  v8f;

// ---------- helpers ----------

__device__ __forceinline__ __bf16 f2bf(float f) {
  unsigned u = __builtin_bit_cast(unsigned, f);
  u += 0x7fffu + ((u >> 16) & 1u);          // round-to-nearest-even
  unsigned short h = (unsigned short)(u >> 16);
  return __builtin_bit_cast(__bf16, h);
}

__device__ __forceinline__ float sigmoidf_fast(float x) {
  return 1.0f / (1.0f + __expf(-x));
}

__device__ __forceinline__ float tanhf_fast(float x) {
  x = fminf(fmaxf(x, -15.0f), 15.0f);
  float e = __expf(2.0f * x);
  return (e - 1.0f) / (e + 1.0f);
}

// A fragment: 16x32 bf16 tile (MxK), source row-major with leading dim ld.
// lane<16 row=lane holds K=0..15; lane>=16 holds K=16..31 (matches load_frag_b).
__device__ __forceinline__ v16bf load_frag_a(const __bf16* base, int ld, int k0, int lane) {
  int r = lane & 15;
  int koff = (lane < 16) ? 0 : 16;
  const __bf16* p = base + (size_t)r * ld + k0 + koff;
  v8bf lo = *(const v8bf*)p;
  v8bf hi = *(const v8bf*)(p + 8);
  v16bf o;
#pragma unroll
  for (int i = 0; i < 8; ++i) { o[i] = lo[i]; o[i + 8] = hi[i]; }
  return o;
}

// B fragment: 32x16 (KxN) tile taken from a transposed (NxK) weight matrix.
// lane gives column n; lane<16 holds K=k0..k0+15, lane>=16 holds K=k0+16..k0+31.
__device__ __forceinline__ v16bf load_frag_b(const __bf16* bt, int ldk, int n0, int k0, int lane) {
  int n = n0 + (lane & 15);
  int koff = (lane < 16) ? 0 : 16;
  return *(const v16bf*)(bt + (size_t)n * ldk + k0 + koff);
}

__device__ __forceinline__ v8f wmma_bf16(v16bf a, v16bf b, v8f c) {
  return __builtin_amdgcn_wmma_f32_16x16x32_bf16(false, a, false, b, (short)0, c, false, false);
}

// Device-wide barrier for a small persistent grid (all blocks co-resident).
// Sense-free generation barrier; agent-scope fences make h-buffer stores
// visible (writer side) and invalidate stale lines (reader side).
__device__ __forceinline__ void grid_barrier(int* cnt, int* gen) {
  __builtin_amdgcn_fence(__ATOMIC_RELEASE, "agent");
  __syncthreads();
  if (threadIdx.x == 0) {
    int g = __hip_atomic_load(gen, __ATOMIC_ACQUIRE, __HIP_MEMORY_SCOPE_AGENT);
    if (__hip_atomic_fetch_add(cnt, 1, __ATOMIC_ACQ_REL, __HIP_MEMORY_SCOPE_AGENT) ==
        (int)gridDim.x - 1) {
      __hip_atomic_store(cnt, 0, __ATOMIC_RELAXED, __HIP_MEMORY_SCOPE_AGENT);
      __hip_atomic_store(gen, g + 1, __ATOMIC_RELEASE, __HIP_MEMORY_SCOPE_AGENT);
    } else {
      while (__hip_atomic_load(gen, __ATOMIC_ACQUIRE, __HIP_MEMORY_SCOPE_AGENT) == g) {
        __builtin_amdgcn_s_sleep(1);
      }
    }
  }
  __syncthreads();
  __builtin_amdgcn_fence(__ATOMIC_ACQUIRE, "agent");
}

// ---------- conversion kernels ----------

__global__ void cvt_bf16_kernel(const float* __restrict__ in, __bf16* __restrict__ out, long n) {
  long i = (long)blockIdx.x * blockDim.x + threadIdx.x;
  if (i < n) out[i] = f2bf(in[i]);
}

// in: [K x N] f32 row-major  ->  out: [N x K] bf16 row-major (transposed)
__global__ void transpose_cvt_kernel(const float* __restrict__ in, __bf16* __restrict__ out,
                                     int K, int N) {
  long i = (long)blockIdx.x * blockDim.x + threadIdx.x;
  long total = (long)K * N;
  if (i >= total) return;
  int k = (int)(i / N);
  int n = (int)(i % N);
  out[(size_t)n * K + k] = f2bf(in[i]);
}

// ---------- input-projection GEMM:  C[16384,2048] = A[16384,K]*B[K,2048] + bias ----------
// Bt is [2048 x K] (transposed). One wave per 16x16 tile, 8 waves/block sharing an
// LDS-staged 16xK A-tile. Grid = 16384 blocks of 256 threads.

__global__ __launch_bounds__(256) void gemm_bias_kernel(
    const __bf16* __restrict__ A, const __bf16* __restrict__ Bt,
    const float* __restrict__ bias, float* __restrict__ C, int K) {
  __shared__ __align__(32) __bf16 ldsA[16 * 1024];
  int tid = threadIdx.x;
  int waveId = tid >> 5;
  int lane = tid & 31;
  int mt = blockIdx.x / 16;                 // 1024 m-tiles (M = 16384)
  int nt = (blockIdx.x % 16) * 8 + waveId;  // 128 n-tiles (N = 2048)

  int chunksPerRow = K >> 3;                // v8bf chunks per row
  int totalChunks = 16 * chunksPerRow;
  for (int idx = tid; idx < totalChunks; idx += 256) {
    int row = idx / chunksPerRow;
    int off = idx % chunksPerRow;
    ((v8bf*)ldsA)[idx] = ((const v8bf*)(A + (size_t)(mt * 16 + row) * K))[off];
  }
  __syncthreads();

  const __bf16* brow = Bt + (size_t)(nt * 16 + (lane & 15)) * K;
  v8f acc = {};
  for (int k0 = 0; k0 < K; k0 += 32) {
    __builtin_prefetch(brow + k0 + 128, 0, 1);   // global_prefetch on streamed B
    v16bf a = load_frag_a(ldsA, K, k0, lane);
    v16bf b = load_frag_b(Bt, K, nt * 16, k0, lane);
    acc = wmma_bf16(a, b, acc);
  }

  int col = nt * 16 + (lane & 15);
  float bs = bias[col];
  int rbase = mt * 16 + ((lane >> 4) << 3);
#pragma unroll
  for (int r = 0; r < 8; ++r) {
    C[(size_t)(rbase + r) * 2048 + col] = acc[r] + bs;
  }
}

// ---------- persistent recurrent pass: the full T-step scan in ONE kernel ----------
// Grid = 16 blocks x 256 threads = 128 waves (4 batch-tiles x 32 hidden-tiles),
// all co-resident; a device-wide barrier separates timesteps. Each wave owns a
// fixed (16-batch x 16-hidden) slice: cell state c and carried h stay in
// REGISTERS for all 256 steps. Per step it computes all four gate tiles
// (i,f,g,o at N offsets 0/512/1024/1536) with 4 WMMA accumulator chains so the
// LSTM cell update is resolved in-register. Only the bf16 h vector (needed by
// other WGPs next step) and the layer output are written.

__global__ __launch_bounds__(256) void lstm_pass_kernel(
    const float* __restrict__ xW,        // [B*T, 2048] pre-activations (with bias)
    const __bf16* __restrict__ Ut,       // [2048, 512] transposed recurrent weights
    __bf16* __restrict__ hbufA,          // [64, 512] h ping buffer (zeroed)
    __bf16* __restrict__ hbufB,          // [64, 512] h pong buffer
    const int* __restrict__ lengths,     // [64]
    int fwd,                             // 1 = forward scan, 0 = reverse scan
    float* __restrict__ outF,            // layer-1: f32 output (d_out), else null
    __bf16* __restrict__ outB,           // layer-0: bf16 output (h0 concat), else null
    int dirOff,                          // 0 = forward half, 512 = backward half
    float* __restrict__ hFin,            // [64,512] final h (f32)
    float* __restrict__ cFin,            // [64,512] final c (f32)
    int* __restrict__ bar_cnt, int* __restrict__ bar_gen) {
  __shared__ __align__(32) __bf16 ldsH[16 * 512];
  const int T = 256, K = 512;
  int tid = threadIdx.x;
  int lane = tid & 31;
  int waveId = tid >> 5;
  int mt = blockIdx.x >> 2;                 // batch tile 0..3
  int nh = (blockIdx.x & 3) * 8 + waveId;   // hidden tile 0..31
  int ch = nh * 16 + (lane & 15);           // hidden column 0..511
  int rbase = mt * 16 + ((lane >> 4) << 3); // batch row base

  float creg[8], hreg[8];
  int lenr[8];
#pragma unroll
  for (int r = 0; r < 8; ++r) {
    creg[r] = 0.0f;
    hreg[r] = 0.0f;
    lenr[r] = lengths[rbase + r];
  }

  for (int s = 0; s < T; ++s) {
    int t = fwd ? s : (T - 1 - s);
    const __bf16* hin = (s & 1) ? hbufB : hbufA;
    __bf16* hout = (s & 1) ? hbufA : hbufB;

    // Early-issue this step's additive pre-activations; these global loads
    // overlap with the WMMA K-loop below.
    float z[4][8];
#pragma unroll
    for (int r = 0; r < 8; ++r) {
      size_t zrow = ((size_t)(rbase + r) * T + t) * 2048;
#pragma unroll
      for (int g = 0; g < 4; ++g) z[g][r] = xW[zrow + g * 512 + ch];
    }

    // stage this block's 16-row slice of h into LDS (shared by its 8 waves)
    for (int idx = tid; idx < 16 * (K >> 3); idx += 256) {
      ((v8bf*)ldsH)[idx] = ((const v8bf*)(hin + (size_t)mt * 16 * K))[idx];
    }
    __syncthreads();

    v8f acc0 = {}, acc1 = {}, acc2 = {}, acc3 = {};
#pragma unroll 2
    for (int k0 = 0; k0 < K; k0 += 32) {
      v16bf a = load_frag_a(ldsH, K, k0, lane);
      v16bf b0 = load_frag_b(Ut, K, 0 * 512 + nh * 16, k0, lane);
      v16bf b1 = load_frag_b(Ut, K, 1 * 512 + nh * 16, k0, lane);
      v16bf b2 = load_frag_b(Ut, K, 2 * 512 + nh * 16, k0, lane);
      v16bf b3 = load_frag_b(Ut, K, 3 * 512 + nh * 16, k0, lane);
      acc0 = wmma_bf16(a, b0, acc0);
      acc1 = wmma_bf16(a, b1, acc1);
      acc2 = wmma_bf16(a, b2, acc2);
      acc3 = wmma_bf16(a, b3, acc3);
    }

#pragma unroll
    for (int r = 0; r < 8; ++r) {
      float ig = sigmoidf_fast(acc0[r] + z[0][r]);
      float fg = sigmoidf_fast(acc1[r] + z[1][r]);
      float gg = tanhf_fast(acc2[r] + z[2][r]);
      float og = sigmoidf_fast(acc3[r] + z[3][r]);
      float c_new = fg * creg[r] + ig * gg;
      float h_new = og * tanhf_fast(c_new);
      bool m = t < lenr[r];                 // Keras masking: carry state through
      creg[r] = m ? c_new : creg[r];
      hreg[r] = m ? h_new : hreg[r];
      size_t si = (size_t)(rbase + r) * 512 + ch;
      hout[si] = f2bf(hreg[r]);
      size_t yi = ((size_t)(rbase + r) * T + t) * 1024 + dirOff + ch;
      if (outF) outF[yi] = hreg[r];
      else      outB[yi] = f2bf(hreg[r]);
    }

    grid_barrier(bar_cnt, bar_gen);         // make h visible to all WGPs
  }

  // final carried state (register-resident) -> f32 buffers
#pragma unroll
  for (int r = 0; r < 8; ++r) {
    size_t si = (size_t)(rbase + r) * 512 + ch;
    hFin[si] = hreg[r];
    cFin[si] = creg[r];
  }
}

// ---------- launch ----------

extern "C" void kernel_launch(void* const* d_in, const int* in_sizes, int n_in,
                              void* d_out, int out_size, void* d_ws, size_t ws_size,
                              hipStream_t stream) {
  const int B = 64, T = 256, INSZ = 512, HSZ = 1024, H = 512, G = 2048;
  (void)in_sizes; (void)n_in; (void)out_size; (void)ws_size;

  const float* x   = (const float*)d_in[0];
  const int*   len = (const int*)d_in[1];
  const float* W0f = (const float*)d_in[2];
  const float* U0f = (const float*)d_in[3];
  const float* b0f = (const float*)d_in[4];
  const float* W0b = (const float*)d_in[5];
  const float* U0b = (const float*)d_in[6];
  const float* b0b = (const float*)d_in[7];
  const float* W1f = (const float*)d_in[8];
  const float* U1f = (const float*)d_in[9];
  const float* b1f = (const float*)d_in[10];
  const float* W1b = (const float*)d_in[11];
  const float* U1b = (const float*)d_in[12];
  const float* b1b = (const float*)d_in[13];

  char* ws = (char*)d_ws;
  size_t off = 0;
  auto walloc = [&](size_t bytes) -> void* {
    void* p = ws + off;
    off = (off + bytes + 255) & ~(size_t)255;
    return p;
  };
  __bf16* xbf  = (__bf16*)walloc((size_t)B * T * INSZ * 2);
  __bf16* h0bf = (__bf16*)walloc((size_t)B * T * HSZ * 2);
  float*  xW   = (float*) walloc((size_t)B * T * G * 4);
  __bf16* W0fT = (__bf16*)walloc((size_t)G * INSZ * 2);
  __bf16* W0bT = (__bf16*)walloc((size_t)G * INSZ * 2);
  __bf16* W1fT = (__bf16*)walloc((size_t)G * HSZ * 2);
  __bf16* W1bT = (__bf16*)walloc((size_t)G * HSZ * 2);
  __bf16* U0fT = (__bf16*)walloc((size_t)G * H * 2);
  __bf16* U0bT = (__bf16*)walloc((size_t)G * H * 2);
  __bf16* U1fT = (__bf16*)walloc((size_t)G * H * 2);
  __bf16* U1bT = (__bf16*)walloc((size_t)G * H * 2);
  __bf16* hbfA = (__bf16*)walloc((size_t)B * H * 2);
  __bf16* hbfB = (__bf16*)walloc((size_t)B * H * 2);
  float*  hst  = (float*) walloc((size_t)B * H * 4);
  float*  cst  = (float*) walloc((size_t)B * H * 4);
  int*    bar  = (int*)   walloc(256);
  int* bar_cnt = bar;
  int* bar_gen = bar + 1;

  // --- convert input & weights to bf16 (weights transposed to [N x K]) ---
  {
    long n = (long)B * T * INSZ;
    cvt_bf16_kernel<<<(int)((n + 255) / 256), 256, 0, stream>>>(x, xbf, n);
  }
  auto tconv = [&](const float* in, __bf16* out, int K) {
    long n = (long)K * G;
    transpose_cvt_kernel<<<(int)((n + 255) / 256), 256, 0, stream>>>(in, out, K, G);
  };
  tconv(W0f, W0fT, INSZ); tconv(W0b, W0bT, INSZ);
  tconv(W1f, W1fT, HSZ);  tconv(W1b, W1bT, HSZ);
  tconv(U0f, U0fT, H);    tconv(U0b, U0bT, H);
  tconv(U1f, U1fT, H);    tconv(U1b, U1bT, H);

  // --- one LSTM pass (single direction): big GEMM + one persistent scan kernel ---
  auto run_pass = [&](const __bf16* Abf, int K, const __bf16* Wt, const float* bias,
                      const __bf16* Ut, float* outF, __bf16* outB, int dirOff, bool fwd,
                      float* hFin, float* cFin) {
    gemm_bias_kernel<<<16384, 256, 0, stream>>>(Abf, Wt, bias, xW, K);
    hipMemsetAsync(hbfA, 0, (size_t)B * H * 2, stream);
    hipMemsetAsync(hbfB, 0, (size_t)B * H * 2, stream);
    hipMemsetAsync(bar, 0, 256, stream);
    lstm_pass_kernel<<<16, 256, 0, stream>>>(xW, Ut, hbfA, hbfB, len, fwd ? 1 : 0,
                                             outF, outB, dirOff, hFin, cFin,
                                             bar_cnt, bar_gen);
  };

  float* rnnout = (float*)d_out;
  float* h1out = rnnout + (size_t)B * T * HSZ;
  float* c1out = h1out + (size_t)B * H;

  // layer 0 (input = x), outputs concatenated into h0bf
  run_pass(xbf, INSZ, W0fT, b0f, U0fT, nullptr, h0bf, 0, true, hst, cst);
  run_pass(xbf, INSZ, W0bT, b0b, U0bT, nullptr, h0bf, 512, false, hst, cst);
  // layer 1 (input = h0); forward pass writes final (h1, c1) straight into d_out
  run_pass(h0bf, HSZ, W1fT, b1f, U1fT, rnnout, nullptr, 0, true, h1out, c1out);
  run_pass(h0bf, HSZ, W1bT, b1b, U1bT, rnnout, nullptr, 512, false, hst, cst);
}
